// Model_61675730371209
// MI455X (gfx1250) — matrix operand order repaired and checked by reading, not compiled, not run on hardware
//
#include <hip/hip_runtime.h>
#include <hip/hip_bf16.h>

typedef __attribute__((ext_vector_type(16))) _Float16 v16h;
typedef __attribute__((ext_vector_type(8)))  _Float16 v8h;
typedef __attribute__((ext_vector_type(8)))  float    v8f;
typedef __attribute__((ext_vector_type(4)))  unsigned int u32x4;
typedef __attribute__((ext_vector_type(8)))  int          i32x8;
typedef __attribute__((ext_vector_type(4)))  int          i32x4;

union V16 { v16h v; v8h h[2]; };

#if defined(__gfx1250__) && __has_builtin(__builtin_amdgcn_tensor_load_to_lds)
#define USE_TDM 1
#endif
#if __has_include(<hip/amd_detail/amd_gfx1250_TDM.h>)
#define TDM_6ARG 1
#endif

__device__ __forceinline__ void wait_tensorcnt0() {
#if defined(__gfx1250__)
#if __has_builtin(__builtin_amdgcn_s_wait_tensorcnt)
  __builtin_amdgcn_s_wait_tensorcnt(0);
#else
  asm volatile("s_wait_tensorcnt 0x0" ::: "memory");
#endif
#endif
}

// ---------------- weight transform: W[cout][cin][125] f32 -> Wt[t][Mpad][Kpad] f16 (zero pad)
__global__ void k_wprep(const float* __restrict__ W, _Float16* __restrict__ Wt,
                        int cout, int cin, int Mpad, int Kpad, long total) {
  long idx = (long)blockIdx.x * blockDim.x + threadIdx.x;
  if (idx >= total) return;
  int c = (int)(idx % Kpad);
  int m = (int)((idx / Kpad) % Mpad);
  int t = (int)(idx / ((long)Kpad * Mpad));
  float v = 0.f;
  if (m < cout && c < cin) v = W[((long)m * cin + c) * 125 + t];
  Wt[idx] = (_Float16)v;
}

// ---------------- input transform: x[b][7][64^3] f32 -> act[b][sp][8] f16
__global__ void k_xprep(const float* __restrict__ x, _Float16* __restrict__ act, long total) {
  long idx = (long)blockIdx.x * blockDim.x + threadIdx.x;
  if (idx >= total) return;
  int  c    = (int)(idx & 7);
  long rest = idx >> 3;
  long sp   = rest % 262144;
  long b    = rest / 262144;
  float v = (c < 7) ? x[(b * 7 + c) * 262144 + sp] : 0.f;
  act[idx] = (_Float16)v;
}

__global__ void k_zero(float* p, int n) {
  int i = blockIdx.x * blockDim.x + threadIdx.x;
  if (i < n) p[i] = 0.f;
}

// ---------------- implicit-GEMM conv via WMMA f16
// block: 32 output channels x 256 output positions; wave: 2x2 tiles of 16x16 (4 WMMA / K-step)
// A (weights) staged per-tap into LDS via TDM (tensor_load_to_lds) when available.
__global__ __launch_bounds__(256)
void k_conv_wmma(const _Float16* __restrict__ act, const _Float16* __restrict__ Wt,
                 float* __restrict__ y, float* __restrict__ gsum, float* __restrict__ gsq,
                 int Bn, int Din, int Hin, int Win, int Dout, int Hout, int Wout,
                 int stride, int cin_pad, int Kpad, int cout, int Mpad) {
  __shared__ __align__(16) _Float16 sA[32 * 128];   // 32 rows x Kpad(<=128) halves, 8KB max
  __shared__ float sSum[32], sSq[32];
  const int tid = threadIdx.x;
  if (tid < 32) { sSum[tid] = 0.f; sSq[tid] = 0.f; }
  __syncthreads();

  const int lane    = tid & 31;
  const int wave    = tid >> 5;
  const int halfsel = lane >> 4;
  const int l15     = lane & 15;

  const long SPo  = (long)Dout * Hout * Wout;
  const long Npos = (long)Bn * SPo;
  const int  Mbase = blockIdx.x * 32;

  // two output columns per lane
  long n0 = (long)blockIdx.y * 256 + (long)wave * 32 + l15;
  long n1 = n0 + 16;
  const bool cv0 = (n0 < Npos), cv1 = (n1 < Npos);
  long q0 = cv0 ? n0 : 0,   q1 = cv1 ? n1 : 0;
  const int  b0 = (int)(q0 / SPo);      const int  b1 = (int)(q1 / SPo);
  const long sp0 = q0 % SPo;            const long sp1 = q1 % SPo;
  const int  HW = Hout * Wout;
  const int  zo0 = (int)(sp0 / HW), r0 = (int)(sp0 % HW);
  const int  zo1 = (int)(sp1 / HW), r1 = (int)(sp1 % HW);
  const int  yo0 = r0 / Wout, xo0 = r0 % Wout;
  const int  yo1 = r1 / Wout, xo1 = r1 % Wout;
  const int  z00 = zo0 * stride - 4, y00 = yo0 * stride - 4, x00 = xo0 * stride - 4;
  const int  z10 = zo1 * stride - 4, y10 = yo1 * stride - 4, x10 = xo1 * stride - 4;

  const int nstep = Kpad >> 5;
  const int c0a   = halfsel * 8;

  v8f acc[2][2];
  acc[0][0] = (v8f)0.0f; acc[0][1] = (v8f)0.0f;
  acc[1][0] = (v8f)0.0f; acc[1][1] = (v8f)0.0f;
  const v8h zero8 = (v8h)(_Float16)0.0f;

  int t = 0;
  for (int kz = 0; kz < 5; ++kz) {
    for (int ky = 0; ky < 5; ++ky) {
      for (int kx = 0; kx < 5; ++kx, ++t) {
        // per-column spatial guards for this tap
        const int zi0 = z00 + kz, yi0 = y00 + ky, xi0 = x00 + kx;
        const int zi1 = z10 + kz, yi1 = y10 + ky, xi1 = x10 + kx;
        const bool ok0 = cv0 && ((unsigned)zi0 < (unsigned)Din) &&
                         ((unsigned)yi0 < (unsigned)Hin) && ((unsigned)xi0 < (unsigned)Win);
        const bool ok1 = cv1 && ((unsigned)zi1 < (unsigned)Din) &&
                         ((unsigned)yi1 < (unsigned)Hin) && ((unsigned)xi1 < (unsigned)Win);
        const long base0 = ok0 ? ((((long)b0 * Din + zi0) * Hin + yi0) * (long)Win + xi0) * cin_pad : 0;
        const long base1 = ok1 ? ((((long)b1 * Din + zi1) * Hin + yi1) * (long)Win + xi1) * cin_pad : 0;

        // ---- stage this tap's 32 x Kpad weight tile into LDS ----
        __syncthreads();   // previous tap's LDS reads complete
#if defined(USE_TDM)
        if (wave == 0) {
          const unsigned ldsb = (unsigned)(size_t)(void*)sA;
          const unsigned long long ga =
              (unsigned long long)(size_t)(Wt + ((long)t * Mpad + Mbase) * (long)Kpad);
          const unsigned kp = (unsigned)Kpad;
          u32x4 g0 = { 1u,                                   // count=1 descriptor
                       ldsb,                                 // lds_addr (bytes)
                       (unsigned)(ga & 0xFFFFFFFFu),
                       (unsigned)((ga >> 32) & 0x1FFFFFFu) | (2u << 30) };  // type=2 (image)
          i32x8 g1 = { (int)0x00010000u,                     // data_size=1 (2 bytes/elem)
                       (int)((kp & 0xFFFFu) << 16),          // tensor_dim0[15:0]
                       (int)(32u << 16),                     // tensor_dim0 hi=0 | tensor_dim1=32
                       (int)(kp << 16),                      // tensor_dim1 hi=0 | tile_dim0=Kpad
                       (int)32,                              // tile_dim1=32 | tile_dim2=0
                       (int)kp,                              // tensor_dim0_stride lo
                       (int)(((kp * 32u) & 0xFFFFu) << 16),  // stride hi=0 | dim1_stride lo
                       (int)((kp * 32u) >> 16) };            // dim1_stride hi
          i32x4 gz = { 0, 0, 0, 0 };
#if defined(TDM_6ARG)
          i32x8 z8 = { 0, 0, 0, 0, 0, 0, 0, 0 };
          __builtin_amdgcn_tensor_load_to_lds(g0, g1, gz, gz, z8, 0);
#else
          __builtin_amdgcn_tensor_load_to_lds(g0, g1, gz, gz, 0);
#endif
          wait_tensorcnt0();
        }
#else
        {
          const _Float16* gsrc = Wt + ((long)t * Mpad + Mbase) * (long)Kpad;
          const int chunks = (32 * Kpad) >> 3;
          for (int i = tid; i < chunks; i += 256)
            ((v8h*)sA)[i] = ((const v8h*)gsrc)[i];
        }
#endif
        __syncthreads();   // tile visible to all waves

        // ---- K loop: 4 WMMAs per step ----
        for (int s = 0; s < nstep; ++s) {
          V16 A0, A1, B0, B1;
          const int aoff = l15 * Kpad + s * 32 + c0a;
          A0.h[0] = *(const v8h*)(sA + aoff);
          A0.h[1] = *(const v8h*)(sA + aoff + 16);
          A1.h[0] = *(const v8h*)(sA + aoff + 16 * Kpad);
          A1.h[1] = *(const v8h*)(sA + aoff + 16 * Kpad + 16);
          const int ci0 = s * 32 + halfsel * 8;
          const bool gv0 = (ci0 < cin_pad), gv1 = (ci0 + 16 < cin_pad);
          B0.h[0] = (ok0 && gv0) ? *(const v8h*)(act + base0 + ci0)     : zero8;
          B0.h[1] = (ok0 && gv1) ? *(const v8h*)(act + base0 + ci0 + 16) : zero8;
          B1.h[0] = (ok1 && gv0) ? *(const v8h*)(act + base1 + ci0)     : zero8;
          B1.h[1] = (ok1 && gv1) ? *(const v8h*)(act + base1 + ci0 + 16) : zero8;
          acc[0][0] = __builtin_amdgcn_wmma_f32_16x16x32_f16(false, A0.v, false, B0.v, (short)0, acc[0][0], false, false);
          acc[0][1] = __builtin_amdgcn_wmma_f32_16x16x32_f16(false, A0.v, false, B1.v, (short)0, acc[0][1], false, false);
          acc[1][0] = __builtin_amdgcn_wmma_f32_16x16x32_f16(false, A1.v, false, B0.v, (short)0, acc[1][0], false, false);
          acc[1][1] = __builtin_amdgcn_wmma_f32_16x16x32_f16(false, A1.v, false, B1.v, (short)0, acc[1][1], false, false);
        }
      }
    }
  }

  // ---- epilogue: store y (NCDHW) + per-channel sum / sumsq for batchnorm ----
  #pragma unroll
  for (int tm = 0; tm < 2; ++tm) {
    #pragma unroll
    for (int r = 0; r < 8; ++r) {
      const int ch = Mbase + tm * 16 + halfsel * 8 + r;
      if (cv0 && ch < cout) y[((long)b0 * cout + ch) * SPo + sp0] = acc[tm][0][r];
      if (cv1 && ch < cout) y[((long)b1 * cout + ch) * SPo + sp1] = acc[tm][1][r];
    }
  }
  #pragma unroll
  for (int tm = 0; tm < 2; ++tm) {
    float ps[8], pq[8];
    #pragma unroll
    for (int r = 0; r < 8; ++r) {
      const float v0 = cv0 ? acc[tm][0][r] : 0.f;
      const float v1 = cv1 ? acc[tm][1][r] : 0.f;
      ps[r] = v0 + v1; pq[r] = v0 * v0 + v1 * v1;
    }
    #pragma unroll
    for (int off = 1; off < 16; off <<= 1) {
      #pragma unroll
      for (int r = 0; r < 8; ++r) {
        ps[r] += __shfl_xor(ps[r], off, 32);
        pq[r] += __shfl_xor(pq[r], off, 32);
      }
    }
    if (l15 == 0) {
      #pragma unroll
      for (int r = 0; r < 8; ++r) {
        atomicAdd(&sSum[tm * 16 + halfsel * 8 + r], ps[r]);
        atomicAdd(&sSq [tm * 16 + halfsel * 8 + r], pq[r]);
      }
    }
  }
  __syncthreads();
  if (tid < 32) {
    const int ch = Mbase + tid;
    if (ch < Mpad) { atomicAdd(&gsum[ch], sSum[tid]); atomicAdd(&gsq[ch], sSq[tid]); }
  }
}

// ---------------- batchnorm stats finalize
__global__ void k_bnprep(const float* __restrict__ gsum, const float* __restrict__ gsq,
                         float* __restrict__ mean, float* __restrict__ inv,
                         long Ncount, int nch) {
  int c = blockIdx.x * blockDim.x + threadIdx.x;
  if (c >= nch) return;
  const float m = gsum[c] / (float)Ncount;
  const float v = gsq[c] / (float)Ncount - m * m;
  mean[c] = m;
  inv[c]  = rsqrtf(v + 1e-5f);
}

// ---------------- BN + gated nonlinearity -> next padded channel-last f16 activation
__global__ void k_gate(const float* __restrict__ y, const float* __restrict__ mean,
                       const float* __restrict__ inv, _Float16* __restrict__ act,
                       long SPo, int cout, int m0, int m1, int m2, int cpadn, long total) {
  long idx = (long)blockIdx.x * blockDim.x + threadIdx.x;
  if (idx >= total) return;
  const int  c    = (int)(idx % cpadn);
  const long rest = idx / cpadn;
  const long sp   = rest % SPo;
  const int  b    = (int)(rest / SPo);
  const int  d    = m0 + 3 * m1 + 5 * m2;
  float out = 0.f;
  if (c < d) {
    const float* yb = y + (long)b * cout * SPo + sp;
    const float v = (yb[(long)c * SPo] - mean[c]) * inv[c];
    if (c < m0) {
      out = v > 0.f ? v : 0.f;                     // l=0: relu
    } else {
      int g;
      if (c < m0 + 3 * m1) g = d + (c - m0) / 3;              // l=1 gate
      else                 g = d + m1 + (c - m0 - 3 * m1) / 5; // l=2 gate
      const float gv = (yb[(long)g * SPo] - mean[g]) * inv[g];
      out = v / (1.f + __expf(-gv));               // v * sigmoid(gate)
    }
  }
  act[idx] = (_Float16)out;
}

// ---------------- final stage: BN + relu fused into global average pool
__global__ void k_pool(const float* __restrict__ y, const float* __restrict__ mean,
                       const float* __restrict__ inv, float* __restrict__ pooled) {
  __shared__ float s[128];
  const int bc = blockIdx.x;          // b*512 + c
  const int c  = bc & 511;
  const float* p = y + (long)bc * 4096;
  const float m = mean[c], iv = inv[c];
  float sum = 0.f;
  for (int i = threadIdx.x; i < 4096; i += 128) {
    const float v = (p[i] - m) * iv;
    sum += v > 0.f ? v : 0.f;
  }
  s[threadIdx.x] = sum;
  __syncthreads();
  for (int off = 64; off > 0; off >>= 1) {
    if (threadIdx.x < off) s[threadIdx.x] += s[threadIdx.x + off];
    __syncthreads();
  }
  if (threadIdx.x == 0) pooled[bc] = s[0] * (1.f / 4096.f);
}

// ---------------- final linear: out[b][o] = pooled[b] . linW[o] + linb[o]
__global__ void k_linear(const float* __restrict__ pooled, const float* __restrict__ linW,
                         const float* __restrict__ linb, float* __restrict__ out) {
  int i = blockIdx.x * blockDim.x + threadIdx.x;
  if (i >= 16 * 20) return;
  const int b = i / 20, o = i % 20;
  const float* pv = pooled + b * 512;
  const float* wv = linW + o * 512;
  float s = linb[o];
  for (int c = 0; c < 512; ++c) s += pv[c] * wv[c];
  out[i] = s;
}

static inline size_t align256(size_t x) { return (x + 255) & ~(size_t)255; }

extern "C" void kernel_launch(void* const* d_in, const int* in_sizes, int n_in,
                              void* d_out, int out_size, void* d_ws, size_t ws_size,
                              hipStream_t stream) {
  (void)in_sizes; (void)n_in; (void)out_size; (void)ws_size;
  const float* x = (const float*)d_in[0];
  const float* Wp[8];
  for (int s = 0; s < 8; ++s) Wp[s] = (const float*)d_in[1 + s];
  const float* linW = (const float*)d_in[9];
  const float* linb = (const float*)d_in[10];
  float* out = (float*)d_out;

  static const int cin_[8]  = {7, 20, 20, 47, 47, 96, 96, 120};
  static const int cout_[8] = {24, 24, 56, 56, 114, 114, 144, 512};
  static const int Dins[8]  = {64, 34, 38, 21, 25, 15, 19, 12};
  static const int Douts[8] = {34, 38, 21, 25, 15, 19, 12, 16};
  static const int strd[8]  = {2, 1, 2, 1, 2, 1, 2, 1};
  static const int M0_[8] = {6, 6, 14, 14, 30, 30, 32, 512};
  static const int M1_[8] = {3, 3, 6, 6, 12, 12, 16, 0};
  static const int M2_[8] = {1, 1, 3, 3, 6, 6, 8, 0};

  int Kpad[8], Mpad[8], cpadIn[8];
  for (int s = 0; s < 8; ++s) {
    Kpad[s]   = ((cin_[s] + 31) / 32) * 32;
    Mpad[s]   = ((cout_[s] + 31) / 32) * 32;   // 32-row M tiles
    cpadIn[s] = ((cin_[s] + 7) / 8) * 8;
  }

  // workspace layout
  char* ws = (char*)d_ws;
  size_t wtOff[8], wtHalves = 0;
  for (int s = 0; s < 8; ++s) { wtOff[s] = wtHalves; wtHalves += 125UL * Mpad[s] * Kpad[s]; }
  size_t off = 0;
  _Float16* WtBase = (_Float16*)(ws + off); off += align256(wtHalves * 2);
  _Float16* actA   = (_Float16*)(ws + off); off += align256((size_t)16 * 262144 * 8 * 2);   // 67 MB
  _Float16* actB   = (_Float16*)(ws + off); off += align256((size_t)16 * 39304 * 24 * 2);   // 30 MB
  float*    ybuf   = (float*)(ws + off);    off += align256((size_t)16 * 512 * 4096 * 4);   // 134 MB
  float*    gsum   = (float*)(ws + off);    off += align256(512 * 4);
  float*    gsq    = (float*)(ws + off);    off += align256(512 * 4);
  float*    meanb  = (float*)(ws + off);    off += align256(512 * 4);
  float*    invb   = (float*)(ws + off);    off += align256(512 * 4);
  float*    pooled = (float*)(ws + off);    off += align256(16 * 512 * 4);

  // weight + input transforms
  for (int s = 0; s < 8; ++s) {
    long total = 125L * Mpad[s] * Kpad[s];
    k_wprep<<<dim3((unsigned)((total + 255) / 256)), 256, 0, stream>>>(
        Wp[s], WtBase + wtOff[s], cout_[s], cin_[s], Mpad[s], Kpad[s], total);
  }
  {
    long total = 16L * 262144 * 8;
    k_xprep<<<dim3((unsigned)((total + 255) / 256)), 256, 0, stream>>>(x, actA, total);
  }

  const _Float16* cur = actA;
  for (int s = 0; s < 8; ++s) {
    k_zero<<<dim3(4), 256, 0, stream>>>(gsum, 512);
    k_zero<<<dim3(4), 256, 0, stream>>>(gsq, 512);

    const int  Do = Douts[s], Di = Dins[s];
    const long SPo  = (long)Do * Do * Do;
    const long Npos = 16L * SPo;
    dim3 grid(Mpad[s] / 32, (unsigned)((Npos + 255) / 256));
    k_conv_wmma<<<grid, 256, 0, stream>>>(cur, WtBase + wtOff[s], ybuf, gsum, gsq,
                                          16, Di, Di, Di, Do, Do, Do,
                                          strd[s], cpadIn[s], Kpad[s], cout_[s], Mpad[s]);
    k_bnprep<<<dim3(2), 256, 0, stream>>>(gsum, gsq, meanb, invb, Npos, 512);

    if (s < 7) {
      _Float16* nxt = (s % 2 == 0) ? actB : actA;
      const int  cpadn = cpadIn[s + 1];
      const long total = Npos * cpadn;
      k_gate<<<dim3((unsigned)((total + 255) / 256)), 256, 0, stream>>>(
          ybuf, meanb, invb, nxt, SPo, cout_[s], M0_[s], M1_[s], M2_[s], cpadn, total);
      cur = nxt;
    }
  }

  k_pool<<<dim3(16 * 512), 128, 0, stream>>>(ybuf, meanb, invb, pooled);
  k_linear<<<dim3(2), 256, 0, stream>>>(pooled, linW, linb, out);
}
